// GNNEncoder_35605278883840
// MI455X (gfx1250) — compile-verified, matrix operand order checked by reading, not run on hardware
//
#include <hip/hip_runtime.h>

// ---------------------------------------------------------------------------
// GCN encoder for MI455X (gfx1250, wave32).
//  - GEMMs: v_wmma_f32_16x16x32_bf16; B staged to LDS in fragment order,
//    A loaded as float4 runs (row index clamped -> no divergent guards)
//  - Edge aggregation: one wave32 per edge, float4/lane, f32 atomics (L2-resident)
//  - BatchNorm: atomic sum/sumsq reduction + fused normalize/ReLU
// ---------------------------------------------------------------------------

typedef __attribute__((ext_vector_type(16))) __bf16 v16bf;
typedef __attribute__((ext_vector_type(8)))  float  v8f;

static constexpr int NN  = 100000;   // nodes
static constexpr int EE  = 1600000;  // edges
static constexpr int GG  = 1024;     // graphs
static constexpr int HH  = 128;      // hidden (== DIN)
static constexpr int EMB = 64;
static constexpr int LL  = 3;

// k index inside a 16-element bf16 fragment (A layout, 16x32 tile):
// lane half 0 holds K = {0..7, 16..23}, half 1 holds K = {8..15, 24..31}
__device__ __forceinline__ int kmap(int hf, int i) {
  return (i < 8) ? (hf * 8 + i) : (16 + hf * 8 + (i - 8));
}

// C[M,NC] = relu?(A[M,128] @ B[128,NC] + bias). One wave -> one 16-row tile.
template <int NC>
__launch_bounds__(256)
__global__ void gemm_k128_wmma(const float* __restrict__ A,
                               const float* __restrict__ B,     // [128, NC] row-major
                               const float* __restrict__ bias,  // [NC] or nullptr
                               float* __restrict__ C, int M, int doRelu) {
  constexpr int K  = 128;
  constexpr int KT = 4;        // 4 k-chunks of 32
  constexpr int NT = NC / 16;  // column tiles
  __shared__ __align__(32) __bf16 Blds[NT * KT * 32 * 16];

  const int tid = threadIdx.x;

  // Cooperative: stage B into LDS already permuted into per-lane fragment order.
  for (int idx = tid; idx < NT * KT * 32 * 16; idx += 256) {
    int i    = idx & 15;
    int lane = (idx >> 4) & 31;
    int kc   = (idx >> 9) & 3;
    int t    = idx >> 11;
    int hf   = lane >> 4;
    int n    = t * 16 + (lane & 15);
    int k    = kc * 32 + kmap(hf, i);
    Blds[idx] = (__bf16)B[k * NC + n];
  }
  __syncthreads();

  const int wave = tid >> 5;
  const int lane = tid & 31;
  const int hf   = lane >> 4;
  const int mrow = lane & 15;
  const int tile_row = (blockIdx.x * 8 + wave) * 16;

  // A fragments. Row index CLAMPED (row m of A only feeds row m of D, and
  // out-of-range rows are never stored) -> unconditional vector loads.
  int r = tile_row + mrow;
  if (r > M - 1) r = M - 1;
  const float* arow = A + (size_t)r * K;

  v16bf a[KT];
#pragma unroll
  for (int kc = 0; kc < KT; ++kc) {
    const float* p = arow + kc * 32 + hf * 8;
    float4 q0 = *reinterpret_cast<const float4*>(p);       // K = kc*32 + hf*8 + 0..3
    float4 q1 = *reinterpret_cast<const float4*>(p + 4);   //                 + 4..7
    float4 q2 = *reinterpret_cast<const float4*>(p + 16);  //            +16  + 0..3
    float4 q3 = *reinterpret_cast<const float4*>(p + 20);  //            +16  + 4..7
    a[kc][0]  = (__bf16)q0.x; a[kc][1]  = (__bf16)q0.y;
    a[kc][2]  = (__bf16)q0.z; a[kc][3]  = (__bf16)q0.w;
    a[kc][4]  = (__bf16)q1.x; a[kc][5]  = (__bf16)q1.y;
    a[kc][6]  = (__bf16)q1.z; a[kc][7]  = (__bf16)q1.w;
    a[kc][8]  = (__bf16)q2.x; a[kc][9]  = (__bf16)q2.y;
    a[kc][10] = (__bf16)q2.z; a[kc][11] = (__bf16)q2.w;
    a[kc][12] = (__bf16)q3.x; a[kc][13] = (__bf16)q3.y;
    a[kc][14] = (__bf16)q3.z; a[kc][15] = (__bf16)q3.w;
  }

  const bool full = (tile_row + 16 <= M);  // wave-uniform

#pragma unroll
  for (int t = 0; t < NT; ++t) {
    v8f acc = {0.f, 0.f, 0.f, 0.f, 0.f, 0.f, 0.f, 0.f};
#pragma unroll
    for (int kc = 0; kc < KT; ++kc) {
      v16bf b = *reinterpret_cast<const v16bf*>(&Blds[((t * KT + kc) * 32 + lane) * 16]);
      acc = __builtin_amdgcn_wmma_f32_16x16x32_bf16(false, a[kc], false, b,
                                                    (short)0, acc, false, false);
    }
    const int col = t * 16 + (lane & 15);
    const float bv = bias ? bias[col] : 0.0f;
    float* cc = C + (size_t)tile_row * NC + col;
    if (full) {
#pragma unroll
      for (int rr = 0; rr < 8; ++rr) {
        float v = acc[rr] + bv;
        if (doRelu) v = fmaxf(v, 0.0f);
        cc[(size_t)(rr + hf * 8) * NC] = v;  // D layout: VGPR rr -> M = rr (+8 hi half)
      }
    } else {
#pragma unroll
      for (int rr = 0; rr < 8; ++rr) {
        int row = tile_row + rr + hf * 8;
        if (row < M) {
          float v = acc[rr] + bv;
          if (doRelu) v = fmaxf(v, 0.0f);
          C[(size_t)row * NC + col] = v;
        }
      }
    }
  }
}

__global__ void fzero(float* p, long n) {
  long i = (long)blockIdx.x * 256 + threadIdx.x;
  if (i < n) p[i] = 0.0f;
}

__global__ void fzero4(float4* p, long n4) {
  long i = (long)blockIdx.x * 256 + threadIdx.x;
  if (i < n4) p[i] = make_float4(0.f, 0.f, 0.f, 0.f);
}

__global__ void deg_accum(const int* __restrict__ dst, float* __restrict__ deg, int E) {
  int e = blockIdx.x * 256 + threadIdx.x;
  if (e < E) atomicAdd(&deg[dst[e]], 1.0f);
}

__global__ void deg_finalize(float* __restrict__ d, int n) {  // deg -> (deg+1)^{-1/2}
  int i = blockIdx.x * 256 + threadIdx.x;
  if (i < n) d[i] = rsqrtf(d[i] + 1.0f);
}

// agg = hw * (1/deg) + bias   (self-loop term; edges accumulated on top)
__global__ void init_agg4(const float4* __restrict__ hw, const float* __restrict__ dinv,
                          const float4* __restrict__ bs4, float4* __restrict__ agg, long n4) {
  long i = (long)blockIdx.x * 256 + threadIdx.x;
  if (i >= n4) return;
  int node = (int)(i >> 5);          // 32 float4 per 128-wide row
  int c4   = (int)(i & 31);
  float dv = dinv[node];
  float w  = dv * dv;
  float4 v = hw[i], b = bs4[c4];
  agg[i] = make_float4(v.x * w + b.x, v.y * w + b.y, v.z * w + b.z, v.w * w + b.w);
}

// One wave per edge: lane owns float4 of the 128-wide row.
__global__ void edge_scatter(const int* __restrict__ src, const int* __restrict__ dst,
                             const float* __restrict__ dinv, const float* __restrict__ hw,
                             float* __restrict__ agg, int E) {
  int gid  = blockIdx.x * 256 + threadIdx.x;
  int e    = gid >> 5;
  if (e >= E) return;
  int lane = threadIdx.x & 31;
  int s = src[e], d = dst[e];
  float w = dinv[s] * dinv[d];
  const float4 v = *reinterpret_cast<const float4*>(hw + (size_t)s * HH + lane * 4);
  float* ap = agg + (size_t)d * HH + lane * 4;
  atomicAdd(ap + 0, v.x * w);
  atomicAdd(ap + 1, v.y * w);
  atomicAdd(ap + 2, v.z * w);
  atomicAdd(ap + 3, v.w * w);
}

// stats[c] += sum, stats[128+c] += sumsq  (per-channel over nodes)
__global__ void bn_stats(const float* __restrict__ agg, float* __restrict__ stats, int Nn) {
  int ch = threadIdx.x & 127;
  int rg = threadIdx.x >> 7;  // 0/1: two rows in flight per block
  const int stride = gridDim.x * 2;
  float s = 0.f, s2 = 0.f;
  for (int n = blockIdx.x * 2 + rg; n < Nn; n += stride) {
    __builtin_prefetch(&agg[(size_t)(n + stride) * HH + ch], 0, 0);  // global_prefetch_b8
    float v = agg[(size_t)n * HH + ch];
    s += v; s2 += v * v;
  }
  atomicAdd(&stats[ch], s);
  atomicAdd(&stats[HH + ch], s2);
}

__global__ void bn_apply4(const float4* __restrict__ agg, const float4* __restrict__ stats4,
                          const float4* __restrict__ gamma4, const float4* __restrict__ beta4,
                          float4* __restrict__ h, long n4, float invN) {
  long i = (long)blockIdx.x * 256 + threadIdx.x;
  if (i >= n4) return;
  int c4 = (int)(i & 31);
  float4 sm = stats4[c4];          // sums
  float4 sq = stats4[32 + c4];     // sum of squares
  float4 g  = gamma4[c4];
  float4 bt = beta4[c4];
  float4 v  = agg[i];
  float4 o;
  {
    float mean = sm.x * invN, var = sq.x * invN - mean * mean;
    o.x = fmaxf((v.x - mean) * rsqrtf(var + 1e-5f) * g.x + bt.x, 0.0f);
  }
  {
    float mean = sm.y * invN, var = sq.y * invN - mean * mean;
    o.y = fmaxf((v.y - mean) * rsqrtf(var + 1e-5f) * g.y + bt.y, 0.0f);
  }
  {
    float mean = sm.z * invN, var = sq.z * invN - mean * mean;
    o.z = fmaxf((v.z - mean) * rsqrtf(var + 1e-5f) * g.z + bt.z, 0.0f);
  }
  {
    float mean = sm.w * invN, var = sq.w * invN - mean * mean;
    o.w = fmaxf((v.w - mean) * rsqrtf(var + 1e-5f) * g.w + bt.w, 0.0f);
  }
  h[i] = o;
}

// One wave per node: lane owns float4; lane 0 bumps the graph count.
__global__ void pool_sum(const float* __restrict__ h, const int* __restrict__ batch,
                         float* __restrict__ pooled, float* __restrict__ cnt, int Nn) {
  int gid  = blockIdx.x * 256 + threadIdx.x;
  int node = gid >> 5;
  if (node >= Nn) return;
  int lane = threadIdx.x & 31;
  int g = batch[node];
  const float4 v = *reinterpret_cast<const float4*>(h + (size_t)node * HH + lane * 4);
  float* pp = pooled + (size_t)g * HH + lane * 4;
  atomicAdd(pp + 0, v.x);
  atomicAdd(pp + 1, v.y);
  atomicAdd(pp + 2, v.z);
  atomicAdd(pp + 3, v.w);
  if (lane == 0) atomicAdd(&cnt[g], 1.0f);
}

__global__ void pool_div4(float4* __restrict__ pooled, const float* __restrict__ cnt, int n4) {
  int i = blockIdx.x * 256 + threadIdx.x;
  if (i >= n4) return;
  float inv = 1.0f / fmaxf(cnt[i >> 5], 1.0f);
  float4 v = pooled[i];
  pooled[i] = make_float4(v.x * inv, v.y * inv, v.z * inv, v.w * inv);
}

static inline int cdiv(long a, long b) { return (int)((a + b - 1) / b); }

extern "C" void kernel_launch(void* const* d_in, const int* in_sizes, int n_in,
                              void* d_out, int out_size, void* d_ws, size_t ws_size,
                              hipStream_t stream) {
  const float* x     = (const float*)d_in[0];
  const int*   eidx  = (const int*)d_in[1];   // [2,E]: src = eidx, dst = eidx + E
  const int*   batch = (const int*)d_in[2];
  const float* W_in  = (const float*)d_in[3];
  const float* b_in  = (const float*)d_in[4];
  const float* Ws    = (const float*)d_in[5];  // [L,H,H]
  const float* bs    = (const float*)d_in[6];  // [L,H]
  const float* gam   = (const float*)d_in[7];  // [L,H]
  const float* bet   = (const float*)d_in[8];  // [L,H]
  const float* W1    = (const float*)d_in[9];
  const float* b1    = (const float*)d_in[10];
  const float* W2    = (const float*)d_in[11];
  const float* b2    = (const float*)d_in[12];
  float* out = (float*)d_out;

  const int* src = eidx;
  const int* dst = eidx + EE;

  // ---- workspace carve-up (256B aligned) ----
  char* base = (char*)d_ws;
  size_t off = 0;
  auto carve = [&](size_t bytes) -> float* {
    float* p = (float*)(base + off);
    off = (off + bytes + 255) & ~(size_t)255;
    return p;
  };
  float* dinv   = carve((size_t)NN * 4);
  float* h      = carve((size_t)NN * HH * 4);
  float* hw     = carve((size_t)NN * HH * 4);
  float* agg    = carve((size_t)NN * HH * 4);
  float* stats  = carve((size_t)2 * HH * 4);
  float* pooled = carve((size_t)GG * HH * 4);
  float* cnt    = carve((size_t)GG * 4);
  float* tbuf   = carve((size_t)GG * HH * 4);
  (void)ws_size; (void)n_in; (void)in_sizes; (void)out_size;

  const long NH  = (long)NN * HH;
  const long NH4 = NH / 4;

  // ---- degrees -> dinv ----
  fzero<<<cdiv(NN, 256), 256, 0, stream>>>(dinv, NN);
  deg_accum<<<cdiv(EE, 256), 256, 0, stream>>>(dst, dinv, EE);
  deg_finalize<<<cdiv(NN, 256), 256, 0, stream>>>(dinv, NN);

  // ---- input projection: h = relu(x @ W_in + b_in) ----
  gemm_k128_wmma<HH><<<cdiv(NN, 128), 256, 0, stream>>>(x, W_in, b_in, h, NN, 1);

  // ---- GCN layers ----
  for (int l = 0; l < LL; ++l) {
    gemm_k128_wmma<HH><<<cdiv(NN, 128), 256, 0, stream>>>(h, Ws + (size_t)l * HH * HH,
                                                          nullptr, hw, NN, 0);
    init_agg4<<<cdiv(NH4, 256), 256, 0, stream>>>((const float4*)hw, dinv,
                                                  (const float4*)(bs + (size_t)l * HH),
                                                  (float4*)agg, NH4);
    edge_scatter<<<cdiv((long)EE * 32, 256), 256, 0, stream>>>(src, dst, dinv, hw, agg, EE);
    fzero<<<1, 256, 0, stream>>>(stats, 2 * HH);
    bn_stats<<<1024, 256, 0, stream>>>(agg, stats, NN);
    bn_apply4<<<cdiv(NH4, 256), 256, 0, stream>>>((const float4*)agg, (const float4*)stats,
                                                  (const float4*)(gam + (size_t)l * HH),
                                                  (const float4*)(bet + (size_t)l * HH),
                                                  (float4*)h, NH4, 1.0f / NN);
  }

  // ---- global mean pool ----
  fzero4<<<cdiv((long)GG * HH / 4, 256), 256, 0, stream>>>((float4*)pooled, (long)GG * HH / 4);
  fzero<<<cdiv(GG, 256), 256, 0, stream>>>(cnt, GG);
  pool_sum<<<cdiv((long)NN * 32, 256), 256, 0, stream>>>(h, batch, pooled, cnt, NN);
  pool_div4<<<cdiv((long)GG * HH / 4, 256), 256, 0, stream>>>((float4*)pooled, cnt, GG * HH / 4);

  // ---- MLP head ----
  gemm_k128_wmma<HH><<<cdiv(GG, 128), 256, 0, stream>>>(pooled, W1, b1, tbuf, GG, 1);
  gemm_k128_wmma<EMB><<<cdiv(GG, 128), 256, 0, stream>>>(tbuf, W2, b2, out, GG, 0);
}
